// SparseAttentionPattern_13288628814328
// MI455X (gfx1250) — matrix operand order repaired and checked by reading, not compiled, not run on hardware
//
#include <hip/hip_runtime.h>

typedef __attribute__((ext_vector_type(16))) _Float16 v16h;
typedef __attribute__((ext_vector_type(8)))  _Float16 v8h;
typedef __attribute__((ext_vector_type(8)))  float    v8f;

// ---------------------------------------------------------------------------
// WMMA f16 16x16x32 operand layouts (wave32):
// A (16x32):  lane L holds row M = L&15; element e -> K = (e<8?0:16) + 8*(L>=16) + (e&7)
// B (32x16):  lane L holds col N = L&15; element e -> K = (e<8?0:16) + 8*(L>=16) + (e&7)
// C/D (16x16): lane L holds col N = L&15; VGPR r -> M = r + 8*(L>=16)
// ---------------------------------------------------------------------------

// fp32 memory -> f16 A operand (GEMM on fp32 inputs)
__device__ __forceinline__ v16h load_a16_f32(const float* __restrict__ p,
                                             int kbase, int hi8) {
  const float4* a = reinterpret_cast<const float4*>(p + kbase + hi8);
  const float4* b = reinterpret_cast<const float4*>(p + kbase + 16 + hi8);
  float4 x0 = a[0], x1 = a[1], y0 = b[0], y1 = b[1];
  v16h r;
  r[0]  = (_Float16)x0.x; r[1]  = (_Float16)x0.y;
  r[2]  = (_Float16)x0.z; r[3]  = (_Float16)x0.w;
  r[4]  = (_Float16)x1.x; r[5]  = (_Float16)x1.y;
  r[6]  = (_Float16)x1.z; r[7]  = (_Float16)x1.w;
  r[8]  = (_Float16)y0.x; r[9]  = (_Float16)y0.y;
  r[10] = (_Float16)y0.z; r[11] = (_Float16)y0.w;
  r[12] = (_Float16)y1.x; r[13] = (_Float16)y1.y;
  r[14] = (_Float16)y1.z; r[15] = (_Float16)y1.w;
  return r;
}

// fp32 memory -> f16 B operand (GEMM on fp32 weights)
__device__ __forceinline__ v16h load_b16_f32(const float* __restrict__ p,
                                             int kbase, int hi16) {
  const float4* a = reinterpret_cast<const float4*>(p + kbase + (hi16 >> 1));
  const float4* b = reinterpret_cast<const float4*>(p + kbase + 16 + (hi16 >> 1));
  float4 x0 = a[0], x1 = a[1], x2 = b[0], x3 = b[1];
  v16h r;
  r[0]  = (_Float16)x0.x; r[1]  = (_Float16)x0.y;
  r[2]  = (_Float16)x0.z; r[3]  = (_Float16)x0.w;
  r[4]  = (_Float16)x1.x; r[5]  = (_Float16)x1.y;
  r[6]  = (_Float16)x1.z; r[7]  = (_Float16)x1.w;
  r[8]  = (_Float16)x2.x; r[9]  = (_Float16)x2.y;
  r[10] = (_Float16)x2.z; r[11] = (_Float16)x2.w;
  r[12] = (_Float16)x3.x; r[13] = (_Float16)x3.y;
  r[14] = (_Float16)x3.z; r[15] = (_Float16)x3.w;
  return r;
}

// f16 memory -> f16 A operand: two 16B chunks, no conversion
__device__ __forceinline__ v16h load_a16_h(const _Float16* __restrict__ p,
                                           int kbase, int hi8) {
  v8h lo = *reinterpret_cast<const v8h*>(p + kbase + hi8);
  v8h hi = *reinterpret_cast<const v8h*>(p + kbase + 16 + hi8);
  return __builtin_shufflevector(lo, hi, 0, 1, 2, 3, 4, 5, 6, 7,
                                         8, 9, 10, 11, 12, 13, 14, 15);
}

// f16 memory -> f16 B operand: two 16B chunks, no conversion
__device__ __forceinline__ v16h load_b16_h(const _Float16* __restrict__ p,
                                           int kbase, int hi16) {
  v8h lo = *reinterpret_cast<const v8h*>(p + kbase + (hi16 >> 1));
  v8h hi = *reinterpret_cast<const v8h*>(p + kbase + 16 + (hi16 >> 1));
  return __builtin_shufflevector(lo, hi, 0, 1, 2, 3, 4, 5, 6, 7,
                                         8, 9, 10, 11, 12, 13, 14, 15);
}

#define WMMA_F16(A, B, C) \
  __builtin_amdgcn_wmma_f32_16x16x32_f16(false, (A), false, (B), (short)0, (C), false, false)

// ---------------------------------------------------------------------------
// Y = (X[TxD] * W^T + bias) * outscale    (torch Linear; W is [D x D] row-major)
// mode 0: Yf32[t][n]             row-major TxD, fp32 (final output)
// mode 1: Yf16[h][t][dh]         head-split f16 (Q with outscale=0.125, K)
// mode 2: Yf16[h][dh][t]         head-split transposed f16 (V)
// Block: 256 threads = 8 waves; wave computes 16x64; grid (T/128, D/64)
// ---------------------------------------------------------------------------
__global__ void gemm_xwT_kernel(const float* __restrict__ X,
                                const float* __restrict__ W,
                                const float* __restrict__ bias,
                                float* __restrict__ Yf32,
                                _Float16* __restrict__ Yf16,
                                int T, int D, int mode, float outscale) {
  const int lane = threadIdx.x & 31;
  const int wv   = threadIdx.x >> 5;
  const int lm   = lane & 15;
  const int hi8  = (lane >> 4) << 3;
  const int hi16 = (lane >> 4) << 4;
  const int m0 = blockIdx.x * 128 + wv * 16;
  const int n0 = blockIdx.y * 64;

  const float* xrow = X + (size_t)(m0 + lm) * D;
  v8f c0 = {}, c1 = {}, c2 = {}, c3 = {};

  for (int k0 = 0; k0 < D; k0 += 32) {
    v16h a  = load_a16_f32(xrow, k0, hi8);
    v16h b0 = load_b16_f32(W + (size_t)(n0 +  0 + lm) * D, k0, hi16);
    v16h b1 = load_b16_f32(W + (size_t)(n0 + 16 + lm) * D, k0, hi16);
    v16h b2 = load_b16_f32(W + (size_t)(n0 + 32 + lm) * D, k0, hi16);
    v16h b3 = load_b16_f32(W + (size_t)(n0 + 48 + lm) * D, k0, hi16);
    c0 = WMMA_F16(a, b0, c0);
    c1 = WMMA_F16(a, b1, c1);
    c2 = WMMA_F16(a, b2, c2);
    c3 = WMMA_F16(a, b3, c3);
  }

  v8f acc[4] = {c0, c1, c2, c3};
#pragma unroll
  for (int nt = 0; nt < 4; ++nt) {
    const int n = n0 + nt * 16 + lm;
    const float bv = bias[n];
    const int h = n >> 6, dh = n & 63;
#pragma unroll
    for (int r = 0; r < 8; ++r) {
      const int m = m0 + r + hi8;
      const float val = (acc[nt][r] + bv) * outscale;
      if (mode == 0)      Yf32[(size_t)m * D + n] = val;
      else if (mode == 1) Yf16[((size_t)h * T + m) * 64 + dh] = (_Float16)val;
      else                Yf16[((size_t)h * 64 + dh) * T + m] = (_Float16)val;
    }
  }
}

// ---------------------------------------------------------------------------
// Causal flash attention.  One (head, 128-query block) per 128-thread block;
// each wave owns 32 queries as two 16-row tiles (u = 0,1) so every K/V operand
// load feeds two WMMA chains (2x arithmetic intensity).
// Computes S^T = K * Q^T: the P^T C-layout is lane-aligned with the A-layout
// required by the P*V WMMA, so softmax needs no cross-lane transpose.
// Q:[H][T][64] f16 (pre-scaled)   K:[H][T][64] f16   VT:[H][64][T] f16
// O:[T][512] f32.  Query-block order reversed so heavy blocks launch first.
// ---------------------------------------------------------------------------
__global__ void flash_attn_kernel(const _Float16* __restrict__ Q,
                                  const _Float16* __restrict__ K,
                                  const _Float16* __restrict__ VT,
                                  float* __restrict__ O,
                                  int T) {
  const int lane = threadIdx.x & 31;
  const int wv   = threadIdx.x >> 5;
  const int h    = blockIdx.y;
  const int lm   = lane & 15;
  const int hi8  = (lane >> 4) << 3;
  const int hi16 = (lane >> 4) << 4;
  const int xb   = (int)gridDim.x - 1 - (int)blockIdx.x;  // heavy blocks first
  const int qb0  = xb * 128 + wv * 32;                    // 32 queries / wave

  // Q^T B-tiles for both query tiles (softmax scale folded into Q).
  v16h bq[2][2];
#pragma unroll
  for (int u = 0; u < 2; ++u) {
    const _Float16* qrow = Q + ((size_t)h * T + (qb0 + u * 16 + lm)) * 64;
    bq[u][0] = load_b16_h(qrow,  0, hi16);
    bq[u][1] = load_b16_h(qrow, 32, hi16);
  }

  v8f o[2][4] = {};
  float m[2] = {-1e30f, -1e30f};
  float l[2] = {0.0f, 0.0f};

  const int ntiles = (qb0 >> 5) + 1;   // keys up to qb0+31 inclusive
  for (int j = 0; j < ntiles; ++j) {
    const int kb = j * 32;

    // Shared operands for both query tiles: K A-tiles and V^T B-tiles.
    v16h a[2][2];
#pragma unroll
    for (int t = 0; t < 2; ++t) {
      const _Float16* krow = K + ((size_t)h * T + (kb + t * 16 + lm)) * 64;
      a[t][0] = load_a16_h(krow,  0, hi8);
      a[t][1] = load_a16_h(krow, 32, hi8);
    }
    const _Float16* vb = VT + (size_t)h * 64 * T;
    const v16h bv0 = load_b16_h(vb + (size_t)( 0 + lm) * T, kb, hi16);
    const v16h bv1 = load_b16_h(vb + (size_t)(16 + lm) * T, kb, hi16);
    const v16h bv2 = load_b16_h(vb + (size_t)(32 + lm) * T, kb, hi16);
    const v16h bv3 = load_b16_h(vb + (size_t)(48 + lm) * T, kb, hi16);

#pragma unroll
    for (int u = 0; u < 2; ++u) {
      const int qu = qb0 + u * 16;

      // S^T tiles: 32 keys x 16 queries.  C lane = query (lm), M = local key.
      v8f st[2];
#pragma unroll
      for (int t = 0; t < 2; ++t) {
        v8f s = {};
        s = WMMA_F16(a[t][0], bq[u][0], s);
        s = WMMA_F16(a[t][1], bq[u][1], s);
        if (kb + t * 16 + 15 > qu) {   // tile touches/passes the diagonal
#pragma unroll
          for (int r = 0; r < 8; ++r) {
            if (kb + t * 16 + r + hi8 > qu + lm) s[r] = -1e30f;
          }
        }
        st[t] = s;
      }

      // Online softmax; lanes (lm) / (lm+16) hold complementary key halves.
      float mx = -1e30f;
#pragma unroll
      for (int r = 0; r < 8; ++r) mx = fmaxf(mx, fmaxf(st[0][r], st[1][r]));
      mx = fmaxf(mx, __shfl_xor(mx, 16, 32));
      const float mnew  = fmaxf(m[u], mx);
      const float alpha = __expf(m[u] - mnew);

      v16h pa;               // P in A-layout: lane-aligned with S^T C-layout
      float rs = 0.0f;
#pragma unroll
      for (int r = 0; r < 8; ++r) {
        const float p0 = __expf(st[0][r] - mnew);
        const float p1 = __expf(st[1][r] - mnew);
        rs += p0 + p1;
        pa[r]     = (_Float16)p0;
        pa[r + 8] = (_Float16)p1;
      }
      rs += __shfl_xor(rs, 16, 32);
      l[u] = l[u] * alpha + rs;
      m[u] = mnew;

      // Rescale O: C-row r of this lane is query (r + hi8) -> broadcast alpha.
#pragma unroll
      for (int r = 0; r < 8; ++r) {
        const float ar = __shfl(alpha, r + hi8, 32);
        o[u][0][r] *= ar; o[u][1][r] *= ar;
        o[u][2][r] *= ar; o[u][3][r] *= ar;
      }

      // O += P(16x32) * V_j(32x64)
      o[u][0] = WMMA_F16(pa, bv0, o[u][0]);
      o[u][1] = WMMA_F16(pa, bv1, o[u][1]);
      o[u][2] = WMMA_F16(pa, bv2, o[u][2]);
      o[u][3] = WMMA_F16(pa, bv3, o[u][3]);
    }
  }

#pragma unroll
  for (int u = 0; u < 2; ++u) {
    const float linv = 1.0f / l[u];
#pragma unroll
    for (int r = 0; r < 8; ++r) {
      const float lr = __shfl(linv, r + hi8, 32);
      float* orow = O + (size_t)(qb0 + u * 16 + r + hi8) * 512 + h * 64 + lm;
      orow[0]  = o[u][0][r] * lr;
      orow[16] = o[u][1][r] * lr;
      orow[32] = o[u][2][r] * lr;
      orow[48] = o[u][3][r] * lr;
    }
  }
}

// ---------------------------------------------------------------------------
extern "C" void kernel_launch(void* const* d_in, const int* in_sizes, int n_in,
                              void* d_out, int out_size, void* d_ws, size_t ws_size,
                              hipStream_t stream) {
  const float* x  = (const float*)d_in[0];
  const float* wq = (const float*)d_in[1];
  const float* bq = (const float*)d_in[2];
  const float* wk = (const float*)d_in[3];
  const float* bk = (const float*)d_in[4];
  const float* wv = (const float*)d_in[5];
  const float* bv = (const float*)d_in[6];
  const float* wo = (const float*)d_in[7];
  const float* bo = (const float*)d_in[8];
  float* out = (float*)d_out;

  const int D = 512;
  const int H = 8;
  const int T = in_sizes[0] / D;   // 4096
  const size_t TD = (size_t)T * D;

  // Workspace: 3 f16 tensors (Q, K, V^T) then f32 attention output.
  _Float16* qws  = (_Float16*)d_ws;           // [H][T][64] f16, pre-scaled
  _Float16* kws  = qws + TD;                  // [H][T][64] f16
  _Float16* vtws = qws + 2 * TD;              // [H][64][T] f16
  float*    ows  = (float*)(qws + 4 * TD);    // [T][D] f32 (16B-aligned)
  (void)H; (void)ws_size; (void)n_in; (void)out_size;

  dim3 ggrid(T / 128, D / 64);
  gemm_xwT_kernel<<<ggrid, 256, 0, stream>>>(x, wq, bq, nullptr, qws,  T, D, 1, 0.125f);
  gemm_xwT_kernel<<<ggrid, 256, 0, stream>>>(x, wk, bk, nullptr, kws,  T, D, 1, 1.0f);
  gemm_xwT_kernel<<<ggrid, 256, 0, stream>>>(x, wv, bv, nullptr, vtws, T, D, 2, 1.0f);

  flash_attn_kernel<<<dim3(T / 128, H), 128, 0, stream>>>(qws, kws, vtws, ows, T);

  gemm_xwT_kernel<<<ggrid, 256, 0, stream>>>(ows, wo, bo, out, nullptr, T, D, 0, 1.0f);
}
